// TileRefineCostLookup_49684181680670
// MI455X (gfx1250) — matrix operand order repaired and checked by reading, not compiled, not run on hardware
//
#include <hip/hip_runtime.h>
#include <hip/hip_bf16.h>
#include <cstdint>

// ---------------------------------------------------------------------------
// Problem constants (from reference)
// ---------------------------------------------------------------------------
constexpr int BB = 4, CC = 128, HH = 192, WW = 384;
constexpr int TF = 32, HID = 48, HR = 2, GG = 8;
constexpr int DD = 2 * HR + 1;                   // 5
constexpr int COST_CH = GG * DD;                 // 40
constexpr int IN_CH = 2 * CC + TF + 4 + COST_CH; // 332
constexpr int CP1 = 352;                         // 332 padded to 11*32
constexpr int CCH1 = 11;
constexpr int CP2 = 64;                          // 48 padded to 2*32
constexpr int CCH2 = 2;
constexpr int PW = WW + 2, PH = HH + 2;          // zero-guard padded extent
constexpr int GUARD_PER_B = 2 * PW + 2 * HH;     // 1156 guard pixels per image
constexpr long long NPIX = (long long)BB * HH * WW;  // 294912
constexpr long long HWp  = (long long)HH * WW;       // 73728
constexpr long long BHW  = NPIX;
constexpr size_t PPIX = (size_t)BB * PH * PW;        // padded pixels, 299536

// Workspace layout (bytes)
constexpr size_t SZ_X   = PPIX * CP1 * 2;
constexpr size_t SZ_PRE = (size_t)NPIX * HID * 4;
constexpr size_t SZ_ACT = PPIX * CP2 * 2;
constexpr size_t SZ_W1  = (size_t)3 * 9 * CCH1 * 32 * 16 * 2;
constexpr size_t SZ_W23 = (size_t)3 * 9 * CCH2 * 32 * 16 * 2;
constexpr size_t SZ_WH  = (size_t)3 * 1 * 2 * 32 * 16 * 2;
constexpr size_t OFF_X    = 0;
constexpr size_t OFF_PRE  = OFF_X + SZ_X;
constexpr size_t OFF_ACTA = OFF_PRE + SZ_PRE;
constexpr size_t OFF_ACTB = OFF_ACTA + SZ_ACT;
constexpr size_t OFF_W1   = OFF_ACTB + SZ_ACT;
constexpr size_t OFF_W2   = OFF_W1 + SZ_W1;
constexpr size_t OFF_W3   = OFF_W2 + SZ_W23;
constexpr size_t OFF_WH   = OFF_W3 + SZ_W23;
constexpr size_t OFF_ST   = OFF_WH + SZ_WH;      // 3 layers * 64 f32
constexpr size_t OFF_SS   = OFF_ST + 3 * 64 * 4; // 3 layers * (192+192) f32

// ---------------------------------------------------------------------------
// WMMA types
// ---------------------------------------------------------------------------
typedef __attribute__((ext_vector_type(16))) _Float16 v16h;
typedef __attribute__((ext_vector_type(8)))  float    v8f;

union Frag16 { v16h h; uint4 u[2]; };
union AccU   { v8f v; float f[8]; };

static __device__ inline v8f wmma_f16(v16h a, v16h b, v8f c) {
  return __builtin_amdgcn_wmma_f32_16x16x32_f16(false, a, false, b, (short)0, c,
                                                false, false);
}

// ---------------------------------------------------------------------------
// zero GN statistic accumulators
// ---------------------------------------------------------------------------
__global__ void zeroK(float* p, int n) {
  int i = blockIdx.x * blockDim.x + threadIdx.x;
  if (i < n) p[i] = 0.f;
}

// ---------------------------------------------------------------------------
// zero the guard ring of a padded f16 NHWC buffer
// ---------------------------------------------------------------------------
__global__ void zeroPadK(_Float16* __restrict__ buf, int CP, int total) {
  int id = blockIdx.x * blockDim.x + threadIdx.x;
  if (id >= total) return;
  int c = id % CP;
  int g = id / CP;
  int b = g / GUARD_PER_B;
  int r = g % GUARD_PER_B;
  int y, x;
  if (r < PW)          { y = 0;       x = r; }
  else if (r < 2 * PW) { y = PH - 1;  x = r - PW; }
  else { int t = r - 2 * PW; y = 1 + (t >> 1); x = (t & 1) ? (PW - 1) : 0; }
  buf[((size_t)(b * PH + y) * PW + x) * CP + c] = (_Float16)0.f;
}

// ---------------------------------------------------------------------------
// pack 3x3 conv weights (OIHW f32) -> WMMA B-fragment layout (f16).
// Entry e = (nt*9 + tap)*CCH + kc; lane (n=l&15, hi=l>>4) holds 16 halves
// j -> K = (j&8)*2 + hi*8 + (j&7), channel = kc*32 + K, out = nt*16 + n.
// ---------------------------------------------------------------------------
__global__ void packConvK(const float* __restrict__ w, _Float16* __restrict__ dst,
                          int O, int I, int CCH, int total) {
  int gid = blockIdx.x * blockDim.x + threadIdx.x;
  if (gid >= total) return;
  int lid = gid & 31;
  int e   = gid >> 5;
  int kc  = e % CCH;
  int tap = (e / CCH) % 9;
  int nt  = e / (CCH * 9);
  int n = lid & 15, hi = lid >> 4;
  int o = nt * 16 + n;
  _Float16* d = dst + (size_t)gid * 16;
  for (int j = 0; j < 16; ++j) {
    int ci = kc * 32 + (j & 8) * 2 + hi * 8 + (j & 7);
    float val = (o < O && ci < I) ? w[((size_t)o * I + ci) * 9 + tap] : 0.f;
    d[j] = (_Float16)val;
  }
}

// pack 1x1 head weights (5 tensors -> 36 outputs, pad N48/K64)
__global__ void packHeadK(const float* __restrict__ hd, const float* __restrict__ hsx,
                          const float* __restrict__ hsy, const float* __restrict__ hc,
                          const float* __restrict__ hf, _Float16* __restrict__ dst) {
  int gid = blockIdx.x * blockDim.x + threadIdx.x;
  if (gid >= 3 * 2 * 32) return;
  int lid = gid & 31;
  int e   = gid >> 5;
  int kc  = e & 1;
  int nt  = e >> 1;
  int n = lid & 15, hi = lid >> 4;
  int o = nt * 16 + n;
  _Float16* d = dst + (size_t)gid * 16;
  for (int j = 0; j < 16; ++j) {
    int ci = kc * 32 + (j & 8) * 2 + hi * 8 + (j & 7);
    float val = 0.f;
    if (ci < HID && o < 36) {
      if      (o == 0) val = hd[ci];
      else if (o == 1) val = hsx[ci];
      else if (o == 2) val = hsy[ci];
      else if (o == 3) val = hc[ci];
      else             val = hf[(size_t)(o - 4) * HID + ci];
    }
    d[j] = (_Float16)val;
  }
}

// ---------------------------------------------------------------------------
// sampling + cost volume + assemble 332-ch input -> f16 padded NHWC (stride 352)
// ---------------------------------------------------------------------------
__global__ __launch_bounds__(256)
void assembleK(const float* __restrict__ fL, const float* __restrict__ fR,
               const float* __restrict__ dsp, const float* __restrict__ sxp,
               const float* __restrict__ syp, const float* __restrict__ feat,
               const float* __restrict__ conf, _Float16* __restrict__ xbuf) {
  long long pid = (long long)blockIdx.x * blockDim.x + threadIdx.x;
  if (pid >= NPIX) return;
  int b = (int)(pid / HWp);
  int rem = (int)(pid % HWp);
  int y = rem / WW, x = rem % WW;
  float d2 = dsp[pid];

  int i0[DD], i1[DD];
  float w0[DD], w1[DD];
  for (int k = 0; k < DD; ++k) {
    float xs = (float)x - (d2 + (float)(k - HR));
    xs = fminf(fmaxf(xs, 0.f), (float)(WW - 1));
    float xf = floorf(xs);
    i0[k] = (int)xf;
    i1[k] = min(i0[k] + 1, WW - 1);
    w1[k] = xs - xf;
    w0[k] = 1.f - w1[k];
  }
  float cost[COST_CH];
  for (int t = 0; t < COST_CH; ++t) cost[t] = 0.f;

  const float* fLb   = fL + (size_t)b * CC * HWp + (size_t)y * WW + x;
  const float* fRrow = fR + (size_t)b * CC * HWp + (size_t)y * WW;
  _Float16* xr = xbuf + ((size_t)(b * PH + 1 + y) * PW + 1 + x) * CP1;

  for (int c = 0; c < CC; ++c) {
    float fl = fLb[(size_t)c * HWp];
    xr[c] = (_Float16)fl;
    const float* rr = fRrow + (size_t)c * HWp;
    int g5 = (c >> 4) * DD;
    for (int k = 0; k < DD; ++k) {
      float v = rr[i0[k]] * w0[k] + rr[i1[k]] * w1[k];
      cost[g5 + k] += fl * v;
      if (k == HR) xr[CC + c] = (_Float16)v;   // off == 0 -> fR_w
    }
  }
  const float* fb = feat + (size_t)b * TF * HWp + (size_t)y * WW + x;
  for (int c = 0; c < TF; ++c) xr[2 * CC + c] = (_Float16)fb[(size_t)c * HWp];
  xr[288] = (_Float16)d2;
  xr[289] = (_Float16)sxp[pid];
  xr[290] = (_Float16)syp[pid];
  xr[291] = (_Float16)conf[pid];
  for (int t = 0; t < COST_CH; ++t)
    xr[292 + t] = (_Float16)(cost[t] * (1.f / 16.f));
  for (int t = IN_CH; t < CP1; ++t) xr[t] = (_Float16)0.f;
}

// ---------------------------------------------------------------------------
// 3x3 conv, implicit-GEMM WMMA. One wave = 32 pixels (2 M-subtiles) x 48 outch.
// Branch-free inner loop: guard-padded input, compile-time offsets.
// Emits h_pre (f32, NHWC stride 48) + GN sum/sumsq atomics per (b, group).
// ---------------------------------------------------------------------------
template <int CP, int CCH>
__global__ __launch_bounds__(128)
void convK(const _Float16* __restrict__ in, const _Float16* __restrict__ wpk,
           float* __restrict__ pre, float* __restrict__ stats) {
  __shared__ float ssum[8], ssq[8];
  if (threadIdx.x < 8) { ssum[threadIdx.x] = 0.f; ssq[threadIdx.x] = 0.f; }
  __syncthreads();

  const int lane  = threadIdx.x & 31;
  const int wtile = blockIdx.x * 4 + (threadIdx.x >> 5);
  const int xt = wtile % (WW / 32);
  const int y  = (wtile / (WW / 32)) % HH;
  const int b  = wtile / ((WW / 32) * HH);
  const int hi = lane >> 4, nl = lane & 15;

  AccU acc[2][3];
  for (int s = 0; s < 2; ++s)
    for (int nt = 0; nt < 3; ++nt)
      for (int j = 0; j < 8; ++j) acc[s][nt].f[j] = 0.f;

  // per-lane base at center tap, subtile 0 (guard ring makes halo in-bounds)
  const _Float16* abase =
      in + ((size_t)(b * PH + 1 + y) * PW + 1 + xt * 32 + nl) * CP + hi * 8;
  const _Float16* qbase = wpk + (size_t)lane * 16;  // lane stride 32B

  for (int ky = 0; ky < 3; ++ky) {
    for (int kx = 0; kx < 3; ++kx) {
      const int tap = ky * 3 + kx;
      const _Float16* ap = abase + ((long long)(ky - 1) * PW + (kx - 1)) * CP;
      const _Float16* qp = qbase + (size_t)tap * CCH * 512;
#pragma unroll
      for (int kc = 0; kc < CCH; ++kc) {
        Frag16 a0, a1;
        a0.u[0] = *(const uint4*)(ap + kc * 32);
        a0.u[1] = *(const uint4*)(ap + kc * 32 + 16);
        a1.u[0] = *(const uint4*)(ap + 16 * CP + kc * 32);
        a1.u[1] = *(const uint4*)(ap + 16 * CP + kc * 32 + 16);
#pragma unroll
        for (int nt = 0; nt < 3; ++nt) {
          Frag16 bw;
          bw.u[0] = *(const uint4*)(qp + ((size_t)nt * 9 * CCH + kc) * 512);
          bw.u[1] = *(const uint4*)(qp + ((size_t)nt * 9 * CCH + kc) * 512 + 8);
          acc[0][nt].v = wmma_f16(a0.h, bw.h, acc[0][nt].v);
          acc[1][nt].v = wmma_f16(a1.h, bw.h, acc[1][nt].v);
        }
      }
    }
  }

  // epilogue: store pre-GN activations (unpadded NHWC), accumulate group stats
  const size_t px0 = ((size_t)b * HH + y) * WW + xt * 32;
#pragma unroll
  for (int s = 0; s < 2; ++s) {
    for (int nt = 0; nt < 3; ++nt) {
      const int ch = nt * 16 + nl;
      float sm = 0.f, q = 0.f;
#pragma unroll
      for (int r = 0; r < 8; ++r) {
        const int m = r + 8 * hi;
        const float v = acc[s][nt].f[r];
        pre[(px0 + s * 16 + m) * HID + ch] = v;
        sm += v;
        q += v * v;
      }
      atomicAdd(&ssum[ch / 6], sm);
      atomicAdd(&ssq[ch / 6], q);
    }
  }
  __syncthreads();
  if (threadIdx.x < 8) {
    atomicAdd(&stats[(b * 8 + threadIdx.x) * 2 + 0], ssum[threadIdx.x]);
    atomicAdd(&stats[(b * 8 + threadIdx.x) * 2 + 1], ssq[threadIdx.x]);
  }
}

// ---------------------------------------------------------------------------
// GN stats -> per (b, channel) scale/shift
// ---------------------------------------------------------------------------
__global__ void gnstatsK(const float* __restrict__ stats, const float* __restrict__ gw,
                         const float* __restrict__ gb, float* __restrict__ scale,
                         float* __restrict__ shift) {
  int t = threadIdx.x;
  if (t >= BB * GG) return;
  int b = t >> 3, g = t & 7;
  float n  = 6.f * HH * WW;
  float su = stats[t * 2], sq = stats[t * 2 + 1];
  float mu = su / n;
  float var = sq / n - mu * mu;
  float rinv = rsqrtf(var + 1e-5f);
  for (int k = 0; k < 6; ++k) {
    int ch = g * 6 + k;
    float sc = gw[ch] * rinv;
    scale[b * HID + ch] = sc;
    shift[b * HID + ch] = gb[ch] - mu * sc;
  }
}

// ---------------------------------------------------------------------------
// apply GN + SiLU, repack f16 padded NHWC (stride 64, ch 48..63 = 0)
// ---------------------------------------------------------------------------
__global__ __launch_bounds__(256)
void actK(const float* __restrict__ pre, const float* __restrict__ scale,
          const float* __restrict__ shift, _Float16* __restrict__ act) {
  long long id = (long long)blockIdx.x * blockDim.x + threadIdx.x;
  if (id >= NPIX * CP2) return;
  long long pix = id >> 6;
  int c = (int)(id & 63);
  int b = (int)(pix / HWp);
  int rem = (int)(pix % HWp);
  int y = rem / WW, x = rem % WW;
  size_t pp = ((size_t)(b * PH + 1 + y) * PW + 1 + x) * CP2;
  if (c < HID) {
    float v = pre[(size_t)pix * HID + c] * scale[b * HID + c] + shift[b * HID + c];
    v = v / (1.f + expf(-v));   // SiLU
    act[pp + c] = (_Float16)v;
  } else {
    act[pp + c] = (_Float16)0.f;
  }
}

// ---------------------------------------------------------------------------
// 1x1 heads as WMMA GEMM (K=48 pad 64, N=36 pad 48) + activation epilogue
// ---------------------------------------------------------------------------
__global__ __launch_bounds__(128)
void headK(const _Float16* __restrict__ act, const _Float16* __restrict__ wpk,
           const float* __restrict__ dsp, const float* __restrict__ sxp,
           const float* __restrict__ syp, const float* __restrict__ conf,
           const float* __restrict__ feat, const float* __restrict__ hd_b,
           const float* __restrict__ hsx_b, const float* __restrict__ hsy_b,
           const float* __restrict__ hc_b, const float* __restrict__ hf_b,
           float* __restrict__ out) {
  const int lane = threadIdx.x & 31;
  const int tile = blockIdx.x * 4 + (threadIdx.x >> 5);
  const int xt = tile % (WW / 16);
  const int y  = (tile / (WW / 16)) % HH;
  const int b  = tile / ((WW / 16) * HH);
  const int hi = lane >> 4, nl = lane & 15;

  AccU acc[3];
  for (int nt = 0; nt < 3; ++nt)
    for (int j = 0; j < 8; ++j) acc[nt].f[j] = 0.f;

  const _Float16* abase =
      act + ((size_t)(b * PH + 1 + y) * PW + 1 + xt * 16 + nl) * CP2 + hi * 8;
  const _Float16* qbase = wpk + (size_t)lane * 16;
#pragma unroll
  for (int kc = 0; kc < CCH2; ++kc) {
    Frag16 a;
    a.u[0] = *(const uint4*)(abase + kc * 32);
    a.u[1] = *(const uint4*)(abase + kc * 32 + 16);
#pragma unroll
    for (int nt = 0; nt < 3; ++nt) {
      Frag16 bw;
      bw.u[0] = *(const uint4*)(qbase + ((size_t)nt * 2 + kc) * 512);
      bw.u[1] = *(const uint4*)(qbase + ((size_t)nt * 2 + kc) * 512 + 8);
      acc[nt].v = wmma_f16(a.h, bw.h, acc[nt].v);
    }
  }

  for (int nt = 0; nt < 3; ++nt) {
    const int n = nt * 16 + nl;
    if (n >= 36) continue;
#pragma unroll
    for (int r = 0; r < 8; ++r) {
      const int m  = r + 8 * hi;
      const int xx = xt * 16 + m;
      const size_t px = ((size_t)b * HH + y) * WW + xx;
      const float v = acc[nt].f[r];
      if (n == 0) {                       // d_out = softplus(conv + d)
        float z = v + hd_b[0] + dsp[px];
        out[px] = (z > 20.f) ? z : log1pf(expf(z));
      } else if (n == 1) {                // sx_out = sx + conv*0.1
        out[BHW + px] = sxp[px] + (v + hsx_b[0]) * 0.1f;
      } else if (n == 2) {                // sy_out = sy + conv*0.1
        out[2 * BHW + px] = syp[px] + (v + hsy_b[0]) * 0.1f;
      } else if (n == 3) {                // conf_out = sigmoid(conv + 2c - 1)
        float z = v + hc_b[0] + 2.f * conf[px] - 1.f;
        out[35 * BHW + px] = 1.f / (1.f + expf(-z));
      } else {                            // feat_out = feat + conv
        int f = n - 4;
        size_t fi = ((size_t)(b * TF + f) * HH + y) * WW + xx;
        out[3 * BHW + fi] = feat[fi] + v + hf_b[f];
      }
    }
  }
}

// ---------------------------------------------------------------------------
// Launcher
// ---------------------------------------------------------------------------
extern "C" void kernel_launch(void* const* d_in, const int* in_sizes, int n_in,
                              void* d_out, int out_size, void* d_ws, size_t ws_size,
                              hipStream_t stream) {
  (void)in_sizes; (void)n_in; (void)out_size; (void)ws_size;
  const float* fL      = (const float*)d_in[0];
  const float* fR      = (const float*)d_in[1];
  const float* dsp     = (const float*)d_in[2];
  const float* sxp     = (const float*)d_in[3];
  const float* syp     = (const float*)d_in[4];
  const float* feat    = (const float*)d_in[5];
  const float* conf    = (const float*)d_in[6];
  const float* conv1_w = (const float*)d_in[7];
  const float* gn1_w   = (const float*)d_in[8];
  const float* gn1_b   = (const float*)d_in[9];
  const float* conv2_w = (const float*)d_in[10];
  const float* gn2_w   = (const float*)d_in[11];
  const float* gn2_b   = (const float*)d_in[12];
  const float* conv3_w = (const float*)d_in[13];
  const float* gn3_w   = (const float*)d_in[14];
  const float* gn3_b   = (const float*)d_in[15];
  const float* hd_w    = (const float*)d_in[16];
  const float* hd_b    = (const float*)d_in[17];
  const float* hsx_w   = (const float*)d_in[18];
  const float* hsx_b   = (const float*)d_in[19];
  const float* hsy_w   = (const float*)d_in[20];
  const float* hsy_b   = (const float*)d_in[21];
  const float* hc_w    = (const float*)d_in[22];
  const float* hc_b    = (const float*)d_in[23];
  const float* hf_w    = (const float*)d_in[24];
  const float* hf_b    = (const float*)d_in[25];

  char* ws = (char*)d_ws;
  _Float16* xbuf = (_Float16*)(ws + OFF_X);
  float*    pre  = (float*)(ws + OFF_PRE);
  _Float16* actA = (_Float16*)(ws + OFF_ACTA);
  _Float16* actB = (_Float16*)(ws + OFF_ACTB);
  _Float16* w1p  = (_Float16*)(ws + OFF_W1);
  _Float16* w2p  = (_Float16*)(ws + OFF_W2);
  _Float16* w3p  = (_Float16*)(ws + OFF_W3);
  _Float16* whp  = (_Float16*)(ws + OFF_WH);
  float* stats   = (float*)(ws + OFF_ST);
  float* stats1 = stats, *stats2 = stats + 64, *stats3 = stats + 128;
  float* ss = (float*)(ws + OFF_SS);
  float* scale1 = ss + 0 * 384, *shift1 = scale1 + 192;
  float* scale2 = ss + 1 * 384, *shift2 = scale2 + 192;
  float* scale3 = ss + 2 * 384, *shift3 = scale3 + 192;
  float* out = (float*)d_out;

  const int convBlocks = BB * HH * (WW / 32) / 4;  // 2304 (2 subtiles/wave)
  const int headBlocks = BB * HH * (WW / 16) / 4;  // 4608

  zeroK<<<1, 256, 0, stream>>>(stats, 3 * 64);
  {
    int gx = BB * GUARD_PER_B * CP1;   // xbuf guard ring
    zeroPadK<<<(gx + 255) / 256, 256, 0, stream>>>(xbuf, CP1, gx);
    int ga = BB * GUARD_PER_B * CP2;   // act guard rings
    zeroPadK<<<(ga + 255) / 256, 256, 0, stream>>>(actA, CP2, ga);
    zeroPadK<<<(ga + 255) / 256, 256, 0, stream>>>(actB, CP2, ga);
  }
  {
    int tot1 = 3 * 9 * CCH1 * 32;  // 9504
    packConvK<<<(tot1 + 255) / 256, 256, 0, stream>>>(conv1_w, w1p, HID, IN_CH, CCH1, tot1);
    int tot2 = 3 * 9 * CCH2 * 32;  // 1728
    packConvK<<<(tot2 + 255) / 256, 256, 0, stream>>>(conv2_w, w2p, HID, HID, CCH2, tot2);
    packConvK<<<(tot2 + 255) / 256, 256, 0, stream>>>(conv3_w, w3p, HID, HID, CCH2, tot2);
    packHeadK<<<1, 192, 0, stream>>>(hd_w, hsx_w, hsy_w, hc_w, hf_w, whp);
  }
  assembleK<<<(int)(NPIX / 256), 256, 0, stream>>>(fL, fR, dsp, sxp, syp, feat, conf, xbuf);

  convK<CP1, CCH1><<<convBlocks, 128, 0, stream>>>(xbuf, w1p, pre, stats1);
  gnstatsK<<<1, 32, 0, stream>>>(stats1, gn1_w, gn1_b, scale1, shift1);
  actK<<<(int)(NPIX * CP2 / 256), 256, 0, stream>>>(pre, scale1, shift1, actA);

  convK<CP2, CCH2><<<convBlocks, 128, 0, stream>>>(actA, w2p, pre, stats2);
  gnstatsK<<<1, 32, 0, stream>>>(stats2, gn2_w, gn2_b, scale2, shift2);
  actK<<<(int)(NPIX * CP2 / 256), 256, 0, stream>>>(pre, scale2, shift2, actB);

  convK<CP2, CCH2><<<convBlocks, 128, 0, stream>>>(actB, w3p, pre, stats3);
  gnstatsK<<<1, 32, 0, stream>>>(stats3, gn3_w, gn3_b, scale3, shift3);
  actK<<<(int)(NPIX * CP2 / 256), 256, 0, stream>>>(pre, scale3, shift3, actA);

  headK<<<headBlocks, 128, 0, stream>>>(actA, whp, dsp, sxp, syp, conf, feat,
                                        hd_b, hsx_b, hsy_b, hc_b, hf_b, out);
}
